// SSM_48026324304222
// MI455X (gfx1250) — matrix operand order repaired and checked by reading, not compiled, not run on hardware
//
#include <hip/hip_runtime.h>

typedef __attribute__((ext_vector_type(16))) _Float16 v16h;
typedef __attribute__((ext_vector_type(8)))  _Float16 v8h;
typedef __attribute__((ext_vector_type(8)))  float    v8f;

namespace {
constexpr int kT = 256, kB = 512, kNX = 256, kNY = 64, kNU = 64, kND = 32;
constexpr int BM   = 16;    // batch rows per workgroup (independent across scan)
constexpr int NTH  = 128;   // 4 wave32 waves
constexpr float HEAT = 1.1591509722222224f;

// LDS layout (units: _Float16)
constexpr int OFF_AW = 0;
constexpr int OFF_BW = OFF_AW + kNX * kNX;   // A_w [NX,NX]
constexpr int OFF_EW = OFF_BW + kNX * kNU;   // B_w [NX,NU]
constexpr int OFF_CW = OFF_EW + kNX * kND;   // E_w [NX,ND]
constexpr int OFF_X0 = OFF_CW + kNY * kNX;   // C_w [NY,NX]
constexpr int OFF_X1 = OFF_X0 + BM * kNX;    // state ping
constexpr int OFF_U  = OFF_X1 + BM * kNX;    // state pong
constexpr int OFF_D  = OFF_U  + BM * kNU;    // u tile
constexpr int SMEM_HALFS = OFF_D + BM * kND; // d tile
constexpr size_t SMEM_BYTES = (size_t)SMEM_HALFS * sizeof(_Float16);
} // namespace

// ---- WMMA fragment helpers (layouts per CDNA5 ISA 7.12.2, wave32) ----

// A-matrix 16x32 f16: lanes 0-15 hold M=0..15; half-wave g=(lane>>4):
//   v16h elems 0..7  = K = g*8 .. g*8+7
//   v16h elems 8..15 = K = 16+g*8 .. 16+g*8+7
__device__ __forceinline__ v16h frag_a(const _Float16* base, int ld, int m0, int k0) {
  const int l = threadIdx.x & 31;
  const _Float16* p = base + (m0 + (l & 15)) * ld + k0 + ((l >> 4) << 3);
  v8h lo = *(const v8h*)p;         // K = kb .. kb+7
  v8h hi = *(const v8h*)(p + 16);  // K = kb+16 .. kb+23
  return __builtin_shufflevector(lo, hi, 0,1,2,3,4,5,6,7,8,9,10,11,12,13,14,15);
}

// B-matrix 32x16 f16: lane n = l&15, elems 0..7 K = (l>>4)*8 + j, elems 8..15 K = 16 + (l>>4)*8 + (j-8).
// B[k][n] == W[n][k] for torch Linear weight W[out,in]  ->  contiguous row read.
__device__ __forceinline__ v16h frag_b(const _Float16* w, int ldk, int n0, int k0) {
  const int l = threadIdx.x & 31;
  const _Float16* p = w + (n0 + (l & 15)) * ldk + k0 + ((l >> 4) << 3);
  v8h lo = *(const v8h*)p;
  v8h hi = *(const v8h*)(p + 16);
  return __builtin_shufflevector(lo, hi, 0,1,2,3,4,5,6,7,8,9,10,11,12,13,14,15);
}

__device__ __forceinline__ v8f wmma(v16h a, v16h b, v8f c) {
  return __builtin_amdgcn_wmma_f32_16x16x32_f16(false, a, false, b, (short)0, c,
                                                false, false);
}

// C/D 16x16 f32: elem i of v8f -> M = (lane>>4)*8 + i, N = lane&15
__device__ __forceinline__ void store_frag_global(float* out, int ldn, int n0, v8f acc) {
  const int l  = threadIdx.x & 31;
  const int n  = n0 + (l & 15);
  const int mb = (l >> 4) << 3;
#pragma unroll
  for (int i = 0; i < 8; ++i) out[(mb + i) * ldn + n] = acc[i];
}

__device__ __forceinline__ void store_frag_lds(_Float16* buf, int ldn, int n0, v8f acc) {
  const int l  = threadIdx.x & 31;
  const int n  = n0 + (l & 15);
  const int mb = (l >> 4) << 3;
#pragma unroll
  for (int i = 0; i < 8; ++i) buf[(mb + i) * ldn + n] = (_Float16)acc[i];
}

__global__ void __launch_bounds__(NTH)
ssm_scan_kernel(const float* __restrict__ x0,  const float* __restrict__ Mf,
                const float* __restrict__ DTp, const float* __restrict__ Dd,
                const float* __restrict__ Aw,  const float* __restrict__ Bw,
                const float* __restrict__ Ew,  const float* __restrict__ Cw,
                const float* __restrict__ x0c,
                float* __restrict__ outX, float* __restrict__ outY,
                float* __restrict__ outU, float* __restrict__ outS) {
  extern __shared__ _Float16 smem[];
  _Float16* lAw = smem + OFF_AW;
  _Float16* lBw = smem + OFF_BW;
  _Float16* lEw = smem + OFF_EW;
  _Float16* lCw = smem + OFF_CW;
  _Float16* lXa = smem + OFF_X0;
  _Float16* lXb = smem + OFF_X1;
  _Float16* lU  = smem + OFF_U;
  _Float16* lD  = smem + OFF_D;

  const int tid  = threadIdx.x;
  const int wave = tid >> 5;
  const int b0   = blockIdx.x * BM;

  // ---- one-time: weights fp32 -> fp16 into LDS; initial state ----
  for (int i = tid; i < kNX * kNX; i += NTH) lAw[i] = (_Float16)Aw[i];
  for (int i = tid; i < kNX * kNU; i += NTH) lBw[i] = (_Float16)Bw[i];
  for (int i = tid; i < kNX * kND; i += NTH) lEw[i] = (_Float16)Ew[i];
  for (int i = tid; i < kNY * kNX; i += NTH) lCw[i] = (_Float16)Cw[i];
  for (int i = tid; i < BM * kNX; i += NTH) {
    int m = i >> 8, n = i & (kNX - 1);
    lXa[i] = (_Float16)(x0[(size_t)(b0 + m) * kNX + n] + x0c[n]);
  }
  if (b0 == 0 && tid == 0) *outS = 0.0f;
  __syncthreads();

  for (int t = 0; t < kT; ++t) {
    const _Float16* xprev = (t & 1) ? lXb : lXa;
    _Float16*       xcur  = (t & 1) ? lXa : lXb;

    const size_t rowU = ((size_t)t * kB + b0) * kNU;
    const size_t rowD = ((size_t)t * kB + b0) * kND;
    const size_t rowX = ((size_t)t * kB + b0) * kNX;
    const size_t rowY = ((size_t)t * kB + b0) * kNY;
    const float* mf = Mf  + rowU;
    const float* dt = DTp + rowU;
    const float* dd = Dd  + rowD;

    // ---- stage u = HEAT*m*dT (store U to global, fp16 copy to LDS) and d ----
#pragma unroll
    for (int i = tid; i < BM * kNU; i += NTH) {
      float u = HEAT * mf[i] * dt[i];
      outU[rowU + i] = u;
      lU[i] = (_Float16)u;
    }
#pragma unroll
    for (int i = tid; i < BM * kND; i += NTH) lD[i] = (_Float16)dd[i];

    if (t + 1 < kT) {  // pull next step's streams toward the WGP
      __builtin_prefetch(mf + (size_t)kB * kNU + tid * 8, 0, 1);
      __builtin_prefetch(dt + (size_t)kB * kNU + tid * 8, 0, 1);
      __builtin_prefetch(dd + (size_t)kB * kND + tid * 4, 0, 1);
    }
    __syncthreads();

    // ---- x_t = x_{t-1} A^T + u B^T + d E^T ----
    // Each wave owns 4 column tiles (64 columns); the state A-fragment is
    // loaded ONCE per K-chunk and shared by 4 back-to-back WMMAs (SRC0 reuse).
    {
      const int nb = wave * 64;
      v8f acc0 = {}, acc1 = {}, acc2 = {}, acc3 = {};
#pragma unroll
      for (int k = 0; k < kNX; k += 32) {
        v16h a = frag_a(xprev, kNX, 0, k);
        acc0 = wmma(a, frag_b(lAw, kNX, nb +  0, k), acc0);
        acc1 = wmma(a, frag_b(lAw, kNX, nb + 16, k), acc1);
        acc2 = wmma(a, frag_b(lAw, kNX, nb + 32, k), acc2);
        acc3 = wmma(a, frag_b(lAw, kNX, nb + 48, k), acc3);
      }
#pragma unroll
      for (int k = 0; k < kNU; k += 32) {
        v16h a = frag_a(lU, kNU, 0, k);
        acc0 = wmma(a, frag_b(lBw, kNU, nb +  0, k), acc0);
        acc1 = wmma(a, frag_b(lBw, kNU, nb + 16, k), acc1);
        acc2 = wmma(a, frag_b(lBw, kNU, nb + 32, k), acc2);
        acc3 = wmma(a, frag_b(lBw, kNU, nb + 48, k), acc3);
      }
      {
        v16h a = frag_a(lD, kND, 0, 0);
        acc0 = wmma(a, frag_b(lEw, kND, nb +  0, 0), acc0);
        acc1 = wmma(a, frag_b(lEw, kND, nb + 16, 0), acc1);
        acc2 = wmma(a, frag_b(lEw, kND, nb + 32, 0), acc2);
        acc3 = wmma(a, frag_b(lEw, kND, nb + 48, 0), acc3);
      }
      store_frag_global(outX + rowX, kNX, nb +  0, acc0);
      store_frag_global(outX + rowX, kNX, nb + 16, acc1);
      store_frag_global(outX + rowX, kNX, nb + 32, acc2);
      store_frag_global(outX + rowX, kNX, nb + 48, acc3);
      store_frag_lds(xcur, kNX, nb +  0, acc0);
      store_frag_lds(xcur, kNX, nb + 16, acc1);
      store_frag_lds(xcur, kNX, nb + 32, acc2);
      store_frag_lds(xcur, kNX, nb + 48, acc3);
    }
    __syncthreads();

    // ---- y_t = x_t C^T  (4 column tiles, 1 per wave) ----
    {
      const int n0 = wave * 16;
      v8f acc = {};
#pragma unroll
      for (int k = 0; k < kNX; k += 32)
        acc = wmma(frag_a(xcur, kNX, 0, k), frag_b(lCw, kNX, n0, k), acc);
      store_frag_global(outY + rowY, kNY, n0, acc);
    }
  }
}

extern "C" void kernel_launch(void* const* d_in, const int* in_sizes, int n_in,
                              void* d_out, int out_size, void* d_ws, size_t ws_size,
                              hipStream_t stream) {
  (void)in_sizes; (void)n_in; (void)out_size; (void)d_ws; (void)ws_size;
  const float* x   = (const float*)d_in[0];
  const float* Mf  = (const float*)d_in[1];
  const float* DT  = (const float*)d_in[2];
  const float* D   = (const float*)d_in[3];
  const float* Aw  = (const float*)d_in[4];
  const float* Bw  = (const float*)d_in[5];
  const float* Ew  = (const float*)d_in[6];
  const float* Cw  = (const float*)d_in[7];
  const float* x0c = (const float*)d_in[8];

  float* outX = (float*)d_out;                         // [T,B,NX]
  float* outY = outX + (size_t)kT * kB * kNX;          // [T,B,NY]
  float* outU = outY + (size_t)kT * kB * kNY;          // [T,B,NU]
  float* outS = outU + (size_t)kT * kB * kNU;          // scalar 0.0

  dim3 grid(kB / BM);   // 32 independent batch-tile scans
  dim3 block(NTH);      // 4 waves
  ssm_scan_kernel<<<grid, block, SMEM_BYTES, stream>>>(
      x, Mf, DT, D, Aw, Bw, Ew, Cw, x0c, outX, outY, outU, outS);
}
